// Softmax_35888746726056
// MI455X (gfx1250) — matrix operand order repaired and checked by reading, not compiled, not run on hardware
//
#include <hip/hip_runtime.h>

// Flash attention, B=2,H=12,S=2048,D=64, fp32 in/out, gfx1250 (wave32).
// Single pass, online softmax (base-2 domain), f16 WMMA w/ f32 accumulate,
// double-buffered LDS tiles with register-staged software pipelining,
// DPP butterfly row-max; row-sum computed on the matrix pipe as P x ones.

typedef __attribute__((ext_vector_type(16))) _Float16 v16h;
typedef __attribute__((ext_vector_type(8)))  float    v8f;

#define BATCH   2
#define NHEADS  12
#define S_LEN   2048
#define DHEAD   64
#define NWAVES  8
#define NTHREADS (32 * NWAVES)        // 256
#define BLOCK_M (16 * NWAVES)         // 128 query rows per workgroup
#define BLOCK_N 64                    // keys per KV tile
#define NTILES  (S_LEN / BLOCK_N)     // 32
#define KSTR    72                    // LDS row stride in halves (144B, 16B-aligned)
#define NEGMIN  (-3.4028234663852886e38f)
#define LOG2E   1.4426950408889634f
#define SCALE2  (0.125f * LOG2E)      // (1/sqrt(64)) * log2(e), base-2 domain

// ---- DPP butterfly max across the 16 lanes of a half-wave ----
template <int CTRL>
__device__ __forceinline__ float dppf(float x) {
    return __builtin_bit_cast(float,
        __builtin_amdgcn_mov_dpp(__builtin_bit_cast(int, x), CTRL, 0xF, 0xF, true));
}
__device__ __forceinline__ float rmax16(float x) {
    x = fmaxf(x, dppf<0xB1>(x));    // quad_perm [1,0,3,2]  (xor 1)
    x = fmaxf(x, dppf<0x4E>(x));    // quad_perm [2,3,0,1]  (xor 2)
    x = fmaxf(x, dppf<0x124>(x));   // row_ror:4
    x = fmaxf(x, dppf<0x128>(x));   // row_ror:8
    return x;
}

__global__ __launch_bounds__(NTHREADS)
void fa_fwd_kernel(const float* __restrict__ q,
                   const float* __restrict__ kmat,
                   const float* __restrict__ vmat,
                   const float* __restrict__ mask,
                   float* __restrict__ out)
{
    __shared__ _Float16 ldsK [2][BLOCK_N][KSTR];   // K tile (key, dim), f16
    __shared__ _Float16 ldsVT[2][DHEAD  ][KSTR];   // V tile transposed (dim, key), f16
    __shared__ _Float16 ldsP [NWAVES][16][KSTR];   // per-wave P staging (row, key)
    __shared__ float    lds_am[S_LEN];             // additive mask row (base-2 units)

    const int tid  = threadIdx.x;
    const int wave = tid >> 5;
    const int lane = tid & 31;
    const int ln   = lane & 15;   // row/col within 16-lane group
    const int g    = lane >> 4;   // K-chunk half held by this lane

    const int bh = blockIdx.y;                      // b*H + h
    const int b  = bh / NHEADS;
    const size_t baseQKV = (size_t)bh * S_LEN * DHEAD;
    const int mbase = blockIdx.x * BLOCK_M;

    const float* kbase = kmat + baseQKV;
    const float* vbase = vmat + baseQKV;

    // ---- Precompute additive mask row (pre-scaled by log2e) into LDS ----
    {
        const float* mrow = mask + (size_t)b * S_LEN;
        #pragma unroll
        for (int i = tid; i < S_LEN; i += NTHREADS)
            lds_am[i] = ((1.0f - mrow[i]) * NEGMIN) * LOG2E;  // 0 or -inf-ish
    }

    // ---- Load Q A-fragments (two 16x32 f16 fragments cover D=64) ----
    // A layout: lane l -> row (l&15); halves 0..7 = K 8g..8g+7, 8..15 = K 16+8g..
    v16h aq[2];
    {
        const int arow = mbase + wave * 16 + ln;
        const float* qp = q + baseQKV + (size_t)arow * DHEAD;
        #pragma unroll
        for (int c = 0; c < 2; ++c) {
            #pragma unroll
            for (int i = 0; i < 8; ++i)
                aq[c][i]     = (_Float16)qp[32 * c + 8 * g + i];
            #pragma unroll
            for (int i = 0; i < 8; ++i)
                aq[c][8 + i] = (_Float16)qp[32 * c + 16 + 8 * g + i];
        }
    }

    // All-ones B fragment: P x ones = per-row sums replicated to all 16 columns
    v16h bones;
    #pragma unroll
    for (int i = 0; i < 16; ++i) bones[i] = (_Float16)1.0f;

    // Online-softmax state in base-2 units.
    // C layout: element v <-> row (v+8g), col (lane&15).
    float mrun[8], lrun[8];
    #pragma unroll
    for (int i = 0; i < 8; ++i) { mrun[i] = -__builtin_inff(); lrun[i] = 0.0f; }
    v8f o[4] = {};   // O accumulator: 16 rows x 64 cols (4 col tiles)

    // Register staging for the next K/V tile: 64x64 f32 over 256 threads = 16 each
    float kreg[16], vreg[16];
    const int idx0 = tid * 16;

    // ---- Prologue: load tile 0 into registers, store to buffer 0 ----
    #pragma unroll
    for (int i = 0; i < 16; ++i) {
        kreg[i] = kbase[idx0 + i];
        vreg[i] = vbase[idx0 + i];
    }
    #pragma unroll
    for (int i = 0; i < 16; ++i) {
        int idx = idx0 + i;
        int kr = idx >> 6, dc = idx & 63;
        ldsK [0][kr][dc] = (_Float16)kreg[i];
        ldsVT[0][dc][kr] = (_Float16)vreg[i];
    }
    __syncthreads();

    for (int t = 0; t < NTILES; ++t) {
        const int p  = t & 1;
        const int kv = t * BLOCK_N;

        // ---- Issue next tile's global loads early (overlap with compute) ----
        if (t + 1 < NTILES) {
            const float* kp = kbase + (size_t)(kv + BLOCK_N) * DHEAD;
            const float* vp = vbase + (size_t)(kv + BLOCK_N) * DHEAD;
            #pragma unroll
            for (int i = 0; i < 16; ++i) {
                kreg[i] = kp[idx0 + i];
                vreg[i] = vp[idx0 + i];
            }
        }

        // ---- S = Q*K^T : four 16x16 f32 fragments (64 keys), K-dim = 64 ----
        // B layout (32x16): lane l -> column (l&15), halves 0..7 -> K 8g+i, 8..15 -> K 16+8g+(i-8).
        v8f sf[4];
        #pragma unroll
        for (int j = 0; j < 4; ++j) {
            v8f c = {};
            #pragma unroll
            for (int cc = 0; cc < 2; ++cc) {
                v16h bk;
                const _Float16* bp = &ldsK[p][16 * j + ln][32 * cc + 8 * g];
                #pragma unroll
                for (int i = 0; i < 16; ++i) bk[i] = bp[i < 8 ? i : i + 8];
                c = __builtin_amdgcn_wmma_f32_16x16x32_f16(
                        false, aq[cc], false, bk, (short)0, c, false, false);
            }
            sf[j] = c;
        }

        // Additive mask per key column (base-2 units, from LDS)
        float am[4];
        #pragma unroll
        for (int j = 0; j < 4; ++j)
            am[j] = lds_am[kv + 16 * j + ln];

        // ---- Online softmax over the 64-key tile, base-2 domain ----
        float corr[8];
        #pragma unroll
        for (int vI = 0; vI < 8; ++vI) {
            float s0 = sf[0][vI] * SCALE2 + am[0];
            float s1 = sf[1][vI] * SCALE2 + am[1];
            float s2 = sf[2][vI] * SCALE2 + am[2];
            float s3 = sf[3][vI] * SCALE2 + am[3];
            float tmax = rmax16(fmaxf(fmaxf(s0, s1), fmaxf(s2, s3)));
            float mnew = fmaxf(mrun[vI], tmax);
            corr[vI] = __builtin_amdgcn_exp2f(mrun[vI] - mnew);
            sf[0][vI] = __builtin_amdgcn_exp2f(s0 - mnew);
            sf[1][vI] = __builtin_amdgcn_exp2f(s1 - mnew);
            sf[2][vI] = __builtin_amdgcn_exp2f(s2 - mnew);
            sf[3][vI] = __builtin_amdgcn_exp2f(s3 - mnew);
            mrun[vI] = mnew;
        }
        #pragma unroll
        for (int f = 0; f < 4; ++f)
            #pragma unroll
            for (int vI = 0; vI < 8; ++vI)
                o[f][vI] *= corr[vI];

        // ---- P: C layout -> A layout via per-wave LDS staging ----
        #pragma unroll
        for (int j = 0; j < 4; ++j)
            #pragma unroll
            for (int vI = 0; vI < 8; ++vI)
                ldsP[wave][vI + 8 * g][16 * j + ln] = (_Float16)sf[j][vI];
        // wave-private region; same-wave DS ops in-order (DScnt waits inserted)
        v16h ap[2];
        #pragma unroll
        for (int kk = 0; kk < 2; ++kk) {
            const _Float16* pp = &ldsP[wave][ln][32 * kk];
            #pragma unroll
            for (int i = 0; i < 8; ++i) ap[kk][i]     = pp[8 * g + i];
            #pragma unroll
            for (int i = 0; i < 8; ++i) ap[kk][8 + i] = pp[16 + 8 * g + i];
        }

        // ---- Row sums on the matrix pipe: rs = P x ones (replicated per col)
        v8f rs = {};
        rs = __builtin_amdgcn_wmma_f32_16x16x32_f16(
                 false, ap[0], false, bones, (short)0, rs, false, false);
        rs = __builtin_amdgcn_wmma_f32_16x16x32_f16(
                 false, ap[1], false, bones, (short)0, rs, false, false);
        #pragma unroll
        for (int vI = 0; vI < 8; ++vI)
            lrun[vI] = lrun[vI] * corr[vI] + rs[vI];

        // ---- O += P * V : 4 col tiles x 2 K-chunks of 32 keys ----
        // B(k,n) = V[key k][dim 16f+n] = ldsVT[16f+n][k], contiguous in k.
        #pragma unroll
        for (int f = 0; f < 4; ++f) {
            v8f acc = o[f];
            #pragma unroll
            for (int kk = 0; kk < 2; ++kk) {
                v16h bv;
                const _Float16* vtp = &ldsVT[p][16 * f + ln][32 * kk + 8 * g];
                #pragma unroll
                for (int i = 0; i < 16; ++i) bv[i] = vtp[i < 8 ? i : i + 8];
                acc = __builtin_amdgcn_wmma_f32_16x16x32_f16(
                          false, ap[kk], false, bv, (short)0, acc, false, false);
            }
            o[f] = acc;
        }

        // ---- Store next tile into the other buffer; one barrier per iter ----
        if (t + 1 < NTILES) {
            #pragma unroll
            for (int i = 0; i < 16; ++i) {
                int idx = idx0 + i;
                int kr = idx >> 6, dc = idx & 63;
                ldsK [1 - p][kr][dc] = (_Float16)kreg[i];
                ldsVT[1 - p][dc][kr] = (_Float16)vreg[i];
            }
        }
        __syncthreads();
    }

    // ---- Normalize and store: row = mbase + wave*16 + v + 8g, col = 16f+ln ----
    float* op = out + baseQKV;
    #pragma unroll
    for (int vI = 0; vI < 8; ++vI) {
        float inv = 1.0f / lrun[vI];
        int row = mbase + wave * 16 + vI + 8 * g;
        #pragma unroll
        for (int f = 0; f < 4; ++f)
            op[(size_t)row * DHEAD + 16 * f + ln] = o[f][vI] * inv;
    }
}

extern "C" void kernel_launch(void* const* d_in, const int* in_sizes, int n_in,
                              void* d_out, int out_size, void* d_ws, size_t ws_size,
                              hipStream_t stream) {
    const float* q    = (const float*)d_in[0];
    const float* k    = (const float*)d_in[1];
    const float* v    = (const float*)d_in[2];
    const float* mask = (const float*)d_in[3];
    float* outp = (float*)d_out;

    dim3 grid(S_LEN / BLOCK_M, BATCH * NHEADS);   // (16, 24)
    dim3 block(NTHREADS);                          // 256 threads = 8 waves
    fa_fwd_kernel<<<grid, block, 0, stream>>>(q, k, v, mask, outp);
}
